// SpatialRefineBlock_39702677684474
// MI455X (gfx1250) — matrix operand order repaired and checked by reading, not compiled, not run on hardware
//
#include <hip/hip_runtime.h>
#include <hip/hip_bf16.h>

// ---------------- problem constants ----------------
#define BB 4
#define TT 32
#define NF 256
#define DD 512
#define HH 8
#define DK 64
#define NTOK (BB*TT*NF)      // 32768 tokens
#define NTW 8                // N-tiles per wave in projection GEMMs (128 cols)

typedef __attribute__((ext_vector_type(16))) _Float16 v16h;
typedef __attribute__((ext_vector_type(8)))  _Float16 v8h;
typedef __attribute__((ext_vector_type(8)))  float    v8f;
typedef __attribute__((ext_vector_type(4)))  float    v4f;

__device__ __forceinline__ v8f wmma16(v16h a, v16h b, v8f c) {
    // D = A(16x32 f16) * B(32x16 f16) + C(16x16 f32)
    return __builtin_amdgcn_wmma_f32_16x16x32_f16(
        /*neg_a=*/false, a, /*neg_b=*/false, b,
        /*c_mod=*/(short)0, c, /*reuse_a=*/false, /*reuse_b=*/false);
}

__device__ __forceinline__ v16h cat8(v8h lo, v8h hi) {
    v16h r;
#pragma unroll
    for (int i = 0; i < 8; ++i) { r[i] = lo[i]; r[i + 8] = hi[i]; }
    return r;
}

// async copy 16B from global to LDS (CDNA5 GLOBAL_LOAD_ASYNC_TO_LDS_B128, ASYNCcnt)
__device__ __forceinline__ void async_g2l_b128(void* lds_dst, const void* gsrc) {
    unsigned lds_addr = (unsigned)(size_t)lds_dst;   // low 32 bits = LDS offset
    asm volatile("global_load_async_to_lds_b128 %0, %1, off"
                 :: "v"(lds_addr), "v"(gsrc) : "memory");
}
__device__ __forceinline__ void wait_async0() {
    asm volatile("s_wait_asynccnt 0" ::: "memory");
}

// ---------------- x -> f16 copy (one shot, removes cvt from GEMM loop) -----
__global__ void xconv_kernel(const float* __restrict__ X, _Float16* __restrict__ Xh) {
    size_t idx = ((size_t)blockIdx.x * blockDim.x + threadIdx.x) * 8;
    v4f a = *(const v4f*)(X + idx);
    v4f b = *(const v4f*)(X + idx + 4);
    v8h r;
#pragma unroll
    for (int i = 0; i < 4; ++i) { r[i] = (_Float16)a[i]; r[i + 4] = (_Float16)b[i]; }
    *(v8h*)(Xh + idx) = r;
}

// ---------------- weight transpose + f16 convert: WT[k][n] = W[n][k] -------
__global__ void wconv_kernel(const float* __restrict__ W, _Float16* __restrict__ WT) {
    int idx = blockIdx.x * blockDim.x + threadIdx.x;   // 512*512 threads
    int n = idx >> 9;
    int k = idx & 511;
    WT[(size_t)k * DD + n] = (_Float16)W[(size_t)n * DD + k];
}

// ---------------- adjacency -> bitmask (256 rows x 8 words) ----------------
__global__ void adjpack_kernel(const int* __restrict__ adj, unsigned* __restrict__ bits) {
    int w = blockIdx.x * blockDim.x + threadIdx.x;     // 2048 words
    int i  = w >> 3;
    int wo = w & 7;
    unsigned b = 0;
#pragma unroll
    for (int jj = 0; jj < 32; ++jj)
        if (adj[(size_t)i * NF + wo * 32 + jj] > 0) b |= (1u << jj);
    bits[w] = b;
}

// ---------------- fused QKV projection -------------------------------------
// Each wave computes one 16-row M tile x 128 columns (8 accumulators).
// Per K-chunk: 1 A fragment (f16 direct load), 8 preloaded B fragments,
// then 8 back-to-back WMMAs (single loadcnt wait per chunk).
__global__ void proj_qkv_kernel(const _Float16* __restrict__ Xh,     // [tok][512] f16
                                const _Float16* __restrict__ WTall,  // 3 x [512][512]
                                const float* __restrict__ bq,
                                const float* __restrict__ bk,
                                const float* __restrict__ bv,
                                _Float16* __restrict__ Qb,   // [bt][h][i][dh]
                                _Float16* __restrict__ KTb,  // [bt][h][dh][j]
                                _Float16* __restrict__ Vb) { // [bt][h][j][dh]
    const int MT = NTOK / 16;        // 2048
    const int NG = DD / (16 * NTW);  // 4 column-groups of 128
    int wave = threadIdx.x >> 5;
    int lane = threadIdx.x & 31;
    int tile = blockIdx.x * (blockDim.x >> 5) + wave;
    if (tile >= 3 * MT * NG) return;
    int proj = tile / (MT * NG);
    int rem  = tile % (MT * NG);
    int m0 = (rem / NG) * 16;
    int c0 = (rem % NG) * (16 * NTW);

    const _Float16* WT = WTall + (size_t)proj * DD * DD;
    const float* bias = (proj == 0) ? bq : (proj == 1) ? bk : bv;

    int kse = lane >> 4;                       // lane-half selects K sub-block
    const _Float16* xr = Xh + (size_t)(m0 + (lane & 15)) * DD;

    v8f acc[NTW];
#pragma unroll
    for (int g = 0; g < NTW; ++g) acc[g] = (v8f){};

    for (int kc = 0; kc < DD; kc += 32) {
        // A fragment: halves 0-7 = K kse*8.., halves 8-15 = +16
        int ka = kc + kse * 8;
        v16h A = cat8(*(const v8h*)(xr + ka), *(const v8h*)(xr + ka + 16));
        // B fragments: lane = N column of B, halves 0-7 = K kse*8.., halves 8-15 = +16
        int kb = kc + kse * 8;
        const _Float16* wrow = WT + (size_t)kb * DD + c0 + (lane & 15);
        __builtin_prefetch(wrow + 32 * DD, 0, 3);
        v16h Bfr[NTW];
#pragma unroll
        for (int g = 0; g < NTW; ++g)
#pragma unroll
            for (int s = 0; s < 16; ++s)
                Bfr[g][s] = wrow[(size_t)(s + (s >= 8 ? 8 : 0)) * DD + g * 16];
#pragma unroll
        for (int g = 0; g < NTW; ++g)
            acc[g] = wmma16(A, Bfr[g], acc[g]);
    }

    int rowoff = kse << 3;
#pragma unroll
    for (int g = 0; g < NTW; ++g) {
        int n  = c0 + g * 16 + (lane & 15);
        int h  = n >> 6;
        int dh = n & 63;
        float bv_ = bias[n];
#pragma unroll
        for (int r = 0; r < 8; ++r) {
            int tok = m0 + r + rowoff;
            int bt = tok >> 8;          // NF = 256
            int i  = tok & 255;
            _Float16 hv = (_Float16)(acc[g][r] + bv_);
            if (proj == 0)
                Qb[(((size_t)bt * HH + h) * NF + i) * DK + dh] = hv;
            else if (proj == 1)
                KTb[(((size_t)bt * HH + h) * DK + dh) * NF + i] = hv;
            else
                Vb[(((size_t)bt * HH + h) * NF + i) * DK + dh] = hv;
        }
    }
}

// ---------------- masked attention per (b,t,h) -----------------------------
// block = 256 threads = 8 waves. KT (64x256) and V (256x64) are staged once
// into LDS with global_load_async_to_lds_b128; each wave then owns 16-row
// slabs of the 256x256 score matrix: S = Qh KT / 8, adjacency mask, softmax,
// probs f16 in LDS, O = P @ V.
__global__ void attn_kernel(const _Float16* __restrict__ Qb,
                            const _Float16* __restrict__ KTb,
                            const _Float16* __restrict__ Vb,
                            const unsigned* __restrict__ adjBits,
                            _Float16* __restrict__ Ob) {    // [bt][i][512]
    extern __shared__ char smem[];
    unsigned* mrow = (unsigned*)smem;                    //  8 KB mask bits
    _Float16* kl = (_Float16*)(smem + NF * 8 * 4);       // 32 KB KT [64][256]
    _Float16* vl = kl + (size_t)DK * NF;                 // 32 KB V  [256][64]
    _Float16* pall = vl + (size_t)NF * DK;               // 64 KB probs

    int bth = blockIdx.x;
    int bt = bth / HH;
    int h  = bth % HH;

    const _Float16* Qh = Qb  + ((size_t)bt * HH + h) * NF * DK;
    const _Float16* KT = KTb + ((size_t)bt * HH + h) * DK * NF;
    const _Float16* Vh = Vb  + ((size_t)bt * HH + h) * NF * DK;
    _Float16* Oh = Ob + (size_t)bt * NF * DD + h * DK;

    // ---- stage KT and V into LDS with async copies (ASYNCcnt) ----
    for (int it = threadIdx.x; it < (DK * NF) / 8; it += blockDim.x)  // 2048 x 16B
        async_g2l_b128(kl + (size_t)it * 8, KT + (size_t)it * 8);
    for (int it = threadIdx.x; it < (NF * DK) / 8; it += blockDim.x)
        async_g2l_b128(vl + (size_t)it * 8, Vh + (size_t)it * 8);
    for (int idx = threadIdx.x; idx < NF * 8; idx += blockDim.x)
        mrow[idx] = adjBits[idx];
    wait_async0();
    __syncthreads();

    int wave = threadIdx.x >> 5;
    int lane = threadIdx.x & 31;
    int kse  = lane >> 4;
    int rowoff = kse << 3;
    _Float16* pw = pall + (size_t)wave * 16 * NF;        // wave-private probs

    for (int slab = wave; slab < 16; slab += 8) {
        int i0 = slab * 16;

        // ---- A fragments of Q (16 x 64, two K-chunks of 32) ----
        const _Float16* qrow = Qh + (size_t)(i0 + (lane & 15)) * DK;
        v16h A0 = cat8(*(const v8h*)(qrow + kse * 8),
                       *(const v8h*)(qrow + kse * 8 + 16));
        v16h A1 = cat8(*(const v8h*)(qrow + 32 + kse * 8),
                       *(const v8h*)(qrow + 32 + kse * 8 + 16));

        // ---- scores: 16 N-tiles x (2 WMMAs), B from LDS ----
        v8f acc[16];
        int kb0 = kse * 8;                               // K index inside chunk
#pragma unroll
        for (int nt = 0; nt < 16; ++nt) {
            const _Float16* k0p = kl + (size_t)kb0 * NF + nt * 16 + (lane & 15);
            const _Float16* k1p = kl + (size_t)(32 + kb0) * NF + nt * 16 + (lane & 15);
            v16h B0, B1;
#pragma unroll
            for (int s = 0; s < 16; ++s) {
                B0[s] = k0p[(size_t)(s + (s >= 8 ? 8 : 0)) * NF];
                B1[s] = k1p[(size_t)(s + (s >= 8 ? 8 : 0)) * NF];
            }
            v8f c = {};
            c = wmma16(A0, B0, c);
            c = wmma16(A1, B1, c);
            acc[nt] = c;
        }

        // ---- mask + row-softmax (row i lives in one 16-lane group) ----
        const float scale = 0.125f;                      // 1/sqrt(64)
#pragma unroll
        for (int r = 0; r < 8; ++r) {
            int i = i0 + r + rowoff;
            unsigned rb[8];
#pragma unroll
            for (int w = 0; w < 8; ++w) rb[w] = mrow[i * 8 + w];
#pragma unroll
            for (int nt = 0; nt < 16; ++nt) {
                int j = nt * 16 + (lane & 15);
                float s = acc[nt][r] * scale;
                acc[nt][r] = ((rb[j >> 5] >> (j & 31)) & 1u) ? s : -1e30f;
            }
            float m = -1e30f;
#pragma unroll
            for (int nt = 0; nt < 16; ++nt) m = fmaxf(m, acc[nt][r]);
#pragma unroll
            for (int d = 1; d < 16; d <<= 1) m = fmaxf(m, __shfl_xor(m, d, 16));
            float s = 0.0f;
#pragma unroll
            for (int nt = 0; nt < 16; ++nt) {
                float e = __expf(acc[nt][r] - m);
                acc[nt][r] = e;
                s += e;
            }
#pragma unroll
            for (int d = 1; d < 16; d <<= 1) s += __shfl_xor(s, d, 16);
            float inv = 1.0f / s;
#pragma unroll
            for (int nt = 0; nt < 16; ++nt) acc[nt][r] *= inv;
        }

        // ---- stage probs (16 x 256 f16, row-major) in LDS ----
#pragma unroll
        for (int nt = 0; nt < 16; ++nt) {
            int j = nt * 16 + (lane & 15);
#pragma unroll
            for (int r = 0; r < 8; ++r)
                pw[(size_t)(r + rowoff) * NF + j] = (_Float16)acc[nt][r];
        }

        // ---- O = P @ V : 4 d-tiles x 8 K-chunks, A and B from LDS ----
        const _Float16* prow = pw + (size_t)(lane & 15) * NF;
#pragma unroll
        for (int dt = 0; dt < 4; ++dt) {
            v8f oc = {};
#pragma unroll
            for (int kc = 0; kc < NF; kc += 32) {
                int ka = kc + kse * 8;
                v16h Af = cat8(*(const v8h*)(prow + ka),
                               *(const v8h*)(prow + ka + 16));
                int kb = kc + kse * 8;                   // j index
                const _Float16* vp = vl + (size_t)kb * DK + dt * 16 + (lane & 15);
                v16h Bf;
#pragma unroll
                for (int s = 0; s < 16; ++s)
                    Bf[s] = vp[(size_t)(s + (s >= 8 ? 8 : 0)) * DK];
                oc = wmma16(Af, Bf, oc);
            }
#pragma unroll
            for (int r = 0; r < 8; ++r) {
                int i  = i0 + r + rowoff;
                int dd = dt * 16 + (lane & 15);
                Oh[(size_t)i * DD + dd] = (_Float16)oc[r];
            }
        }
    }
}

// ---------------- final output projection: out = O @ Wo^T + bo (f32) -------
__global__ void outproj_kernel(const _Float16* __restrict__ Ob,  // [tok][512]
                               const _Float16* __restrict__ WoT, // [k][n]
                               const float* __restrict__ bo,
                               float* __restrict__ out) {        // [tok][512]
    const int MT = NTOK / 16;        // 2048
    const int NG = DD / (16 * NTW);  // 4
    int wave = threadIdx.x >> 5;
    int lane = threadIdx.x & 31;
    int tile = blockIdx.x * (blockDim.x >> 5) + wave;
    if (tile >= MT * NG) return;
    int m0 = (tile / NG) * 16;
    int c0 = (tile % NG) * (16 * NTW);
    int kse = lane >> 4;

    const _Float16* orow = Ob + (size_t)(m0 + (lane & 15)) * DD;
    v8f acc[NTW];
#pragma unroll
    for (int g = 0; g < NTW; ++g) acc[g] = (v8f){};

    for (int kc = 0; kc < DD; kc += 32) {
        int ka = kc + kse * 8;
        v16h A = cat8(*(const v8h*)(orow + ka), *(const v8h*)(orow + ka + 16));
        int kb = kc + kse * 8;
        const _Float16* wrow = WoT + (size_t)kb * DD + c0 + (lane & 15);
        __builtin_prefetch(wrow + 32 * DD, 0, 3);
        v16h Bfr[NTW];
#pragma unroll
        for (int g = 0; g < NTW; ++g)
#pragma unroll
            for (int s = 0; s < 16; ++s)
                Bfr[g][s] = wrow[(size_t)(s + (s >= 8 ? 8 : 0)) * DD + g * 16];
#pragma unroll
        for (int g = 0; g < NTW; ++g)
            acc[g] = wmma16(A, Bfr[g], acc[g]);
    }

    int rowoff = kse << 3;
#pragma unroll
    for (int g = 0; g < NTW; ++g) {
        int n = c0 + g * 16 + (lane & 15);
        float bias = bo[n];
#pragma unroll
        for (int r = 0; r < 8; ++r) {
            int tok = m0 + r + rowoff;
            out[(size_t)tok * DD + n] = acc[g][r] + bias;
        }
    }
}

// ---------------- host-side orchestration ----------------------------------
extern "C" void kernel_launch(void* const* d_in, const int* in_sizes, int n_in,
                              void* d_out, int out_size, void* d_ws, size_t ws_size,
                              hipStream_t stream) {
    const float* x  = (const float*)d_in[0];
    const int*   adj = (const int*)d_in[1];
    const float* Wq = (const float*)d_in[2];
    const float* bq = (const float*)d_in[3];
    const float* Wk = (const float*)d_in[4];
    const float* bk = (const float*)d_in[5];
    const float* Wv = (const float*)d_in[6];
    const float* bv = (const float*)d_in[7];
    const float* Wo = (const float*)d_in[8];
    const float* bo = (const float*)d_in[9];
    float* out = (float*)d_out;

    char* ws = (char*)d_ws;
    const size_t szHalf = (size_t)NTOK * DD * sizeof(_Float16);  // 32 MB
    _Float16* Qb  = (_Float16*)(ws);
    _Float16* KTb = (_Float16*)(ws + szHalf);
    _Float16* Vb  = (_Float16*)(ws + 2 * szHalf);
    _Float16* Ob  = (_Float16*)(ws + 3 * szHalf);
    _Float16* Xh  = Ob;   // aliases O buffer: consumed by proj, before attn writes O
    _Float16* WTq = (_Float16*)(ws + 4 * szHalf);
    _Float16* WTk = WTq + (size_t)DD * DD;
    _Float16* WTv = WTk + (size_t)DD * DD;
    _Float16* WTo = WTv + (size_t)DD * DD;
    unsigned* adjBits = (unsigned*)(ws + 4 * szHalf + 4 * (size_t)DD * DD * sizeof(_Float16));

    // x -> f16 (one shot)
    xconv_kernel<<<(NTOK * (DD / 8)) / 256, 256, 0, stream>>>(x, Xh);

    // weights -> transposed f16
    dim3 wb(256), wg((DD * DD) / 256);
    wconv_kernel<<<wg, wb, 0, stream>>>(Wq, WTq);
    wconv_kernel<<<wg, wb, 0, stream>>>(Wk, WTk);
    wconv_kernel<<<wg, wb, 0, stream>>>(Wv, WTv);
    wconv_kernel<<<wg, wb, 0, stream>>>(Wo, WTo);

    // adjacency -> bitmask
    adjpack_kernel<<<(NF * 8) / 256, 256, 0, stream>>>(adj, adjBits);

    // fused QKV projection (3 * 2048 * 4 wave-tiles of 16x128, 8 waves/block)
    {
        int tiles = 3 * (NTOK / 16) * (DD / (16 * NTW));
        proj_qkv_kernel<<<tiles / 8, 256, 0, stream>>>(Xh, WTq, bq, bk, bv, Qb, KTb, Vb);
    }

    // masked attention, one block per (b,t,h)
    {
        size_t shmem = (size_t)NF * 8 * sizeof(unsigned)        //  8 KB mask
                     + (size_t)DK * NF * sizeof(_Float16)       // 32 KB KT
                     + (size_t)NF * DK * sizeof(_Float16)       // 32 KB V
                     + (size_t)8 * 16 * NF * sizeof(_Float16);  // 64 KB probs
        attn_kernel<<<BB * TT * HH, 256, shmem, stream>>>(Qb, KTb, Vb, adjBits, Ob);
    }

    // output projection (2048 * 4 wave-tiles of 16x128)
    {
        int tiles = (NTOK / 16) * (DD / (16 * NTW));
        outproj_kernel<<<tiles / 8, 256, 0, stream>>>(Ob, WTo, bo, out);
    }
}
